// BertSelfAttention_54726473285713
// MI455X (gfx1250) — 8-bit matrix operand order repaired and checked by reading, compiled, not run on hardware
//
#include <hip/hip_runtime.h>

typedef __attribute__((ext_vector_type(8))) int v8i;

static constexpr int BB = 4, SS = 1024, HH = 1024, NHH = 16, DHH = 64;
static constexpr int MR = BB * SS;
static constexpr float CLIPV = 2.5f;


__global__ void init_scal_k(float* scal) {
  if (threadIdx.x < 8) scal[threadIdx.x] = 0.0f;
}

__global__ void absmax_k(const float* __restrict__ x, int n, unsigned* slot) {
  float m = 0.0f;
  const int stride = gridDim.x * blockDim.x;
  for (int i = blockIdx.x * blockDim.x + threadIdx.x; i < n; i += stride) {
    if (i + stride < n) __builtin_prefetch(x + i + stride, 0, 0);
    m = fmaxf(m, fabsf(x[i]));
  }
#pragma unroll
  for (int off = 16; off > 0; off >>= 1)
    m = fmaxf(m, __shfl_xor(m, off, 32));
  if ((threadIdx.x & 31) == 0) atomicMax(slot, __float_as_uint(m));
}

__global__ void finalize_scales_k(float* scal, int first, int count) {
  int i = threadIdx.x;
  if (i >= count) return;
  float m = scal[first + i];
  m = fminf(m, CLIPV);
  m = fmaxf(m, 1e-8f);
  float s = 127.0f / m;
  scal[8 + first + i]  = s;
  scal[16 + first + i] = 1.0f / s;
}

__global__ void quant8_pack4_k(const float* __restrict__ x, signed char* q,
                               const float* scal, int slot, int n4) {
  int t = blockIdx.x * blockDim.x + threadIdx.x;
  if (t >= n4) return;
  const float s = scal[8 + slot];
  int i = t << 2, r = 0;
#pragma unroll
  for (int j = 0; j < 4; ++j) {
    float v = fminf(fmaxf(x[i + j], -CLIPV), CLIPV);
    r |= (__float2int_rn(v * s) & 0xFF) << (8 * j);
  }
  *reinterpret_cast<int*>(q + i) = r;
}

__global__ void quant_v_k(const float* __restrict__ vf, signed char* vqT,
                          float* vq_out, const float* scal, int n) {
  int i = blockIdx.x * blockDim.x + threadIdx.x;
  if (i >= n) return;
  const float sv = scal[8 + 6], inv = scal[16 + 6];
  int row = i >> 10, col = i & 1023;
  int b = row >> 10, s = row & 1023;
  int h = col >> 6,  d = col & 63;
  float x = fminf(fmaxf(vf[i], -CLIPV), CLIPV);
  int qi = __float2int_rn(x * sv);
  int bh = b * NHH + h;
  vqT[(size_t)bh * (DHH * SS) + d * SS + s] = (signed char)qi;
  vq_out[((size_t)bh * SS + s) * DHH + d]  = (float)qi * inv;
}

__device__ __forceinline__ v8i load_a_frag(const signed char* rowbase, int ld,
                                           int lane) {
  const int m = lane & 15, hi = lane >> 4;
  const signed char* r = rowbase + m * ld;
  v8i a;
#pragma unroll
  for (int v = 0; v < 8; ++v) {
    const int k = ((v >> 1) << 4) + (hi << 3) + ((v & 1) << 2);
    a[v] = *reinterpret_cast<const int*>(r + k);
  }
  return a;
}

__device__ __forceinline__ v8i load_a_frag_quant(const float* rowbase, int ld,
                                                 int lane, float sp) {
  const int m = lane & 15, hi = lane >> 4;
  const float* r = rowbase + m * ld;
  v8i a;
#pragma unroll
  for (int v = 0; v < 8; ++v) {
    const int k = ((v >> 1) << 4) + (hi << 3) + ((v & 1) << 2);
    const float4 f = *reinterpret_cast<const float4*>(r + k);
    int q0 = __float2int_rn(fminf(fmaxf(f.x, -CLIPV), CLIPV) * sp) & 0xFF;
    int q1 = __float2int_rn(fminf(fmaxf(f.y, -CLIPV), CLIPV) * sp) & 0xFF;
    int q2 = __float2int_rn(fminf(fmaxf(f.z, -CLIPV), CLIPV) * sp) & 0xFF;
    int q3 = __float2int_rn(fminf(fmaxf(f.w, -CLIPV), CLIPV) * sp) & 0xFF;
    a[v] = q0 | (q1 << 8) | (q2 << 16) | (q3 << 24);
  }
  return a;
}

__device__ __forceinline__ v8i load_b_frag(const signed char* colbase, int ld,
                                           int lane) {
  const int n = lane & 15, hi = lane >> 4;
  const signed char* c = colbase + n * ld;
  v8i b;
#pragma unroll
  for (int v = 0; v < 8; ++v) {
    const int k = ((v >> 1) << 4) + (hi << 3) + ((v & 1) << 2);
    b[v] = *reinterpret_cast<const int*>(c + k);
  }
  return b;
}

__global__ __launch_bounds__(256) void gemm_qkv_k(
    const signed char* __restrict__ X, const signed char* __restrict__ W,
    const float* __restrict__ bias, const float* __restrict__ scal,
    int sw_slot, float* __restrict__ out) {
  const int wid  = blockIdx.x * (blockDim.x >> 5) + (threadIdx.x >> 5);
  const int lane = threadIdx.x & 31;
  const int tm = wid >> 4;
  const int sn = wid & 15;
  if (tm >= MR / 16) return;

  const signed char* arow  = X + (size_t)(tm * 16) * HH;
  const signed char* wbase = W + (size_t)(sn * 64) * HH;

  v8i acc[4] = {};
  v8i aA = load_a_frag(arow, HH, lane);
  v8i bA[4], bB[4];
#pragma unroll
  for (int t = 0; t < 4; ++t)
    bA[t] = load_b_frag(wbase + (size_t)(t * 16) * HH, HH, lane);

#pragma unroll 1
  for (int kb = 0; kb < HH; kb += 128) {
    v8i aB = load_a_frag(arow + kb + 64, HH, lane);
#pragma unroll
    for (int t = 0; t < 4; ++t)
      bB[t] = load_b_frag(wbase + (size_t)(t * 16) * HH + kb + 64, HH, lane);
#pragma unroll
    for (int t = 0; t < 4; ++t)
      acc[t] = __builtin_amdgcn_wmma_i32_16x16x64_iu8(true, aA, true, bA[t],
                                                      acc[t], false, false);
    const int kn = (kb + 128 < HH) ? (kb + 128) : kb;
    aA = load_a_frag(arow + kn, HH, lane);
#pragma unroll
    for (int t = 0; t < 4; ++t)
      bA[t] = load_b_frag(wbase + (size_t)(t * 16) * HH + kn, HH, lane);
#pragma unroll
    for (int t = 0; t < 4; ++t)
      acc[t] = __builtin_amdgcn_wmma_i32_16x16x64_iu8(true, aB, true, bB[t],
                                                      acc[t], false, false);
  }

  const float sc = scal[16 + 0] * scal[16 + sw_slot];
  const int hi = lane >> 4, nloc = lane & 15;
#pragma unroll
  for (int t = 0; t < 4; ++t) {
    const int ncol = sn * 64 + t * 16 + nloc;
    const float bv = bias[ncol];
#pragma unroll
    for (int r = 0; r < 8; ++r) {
      const int mrow = tm * 16 + r + hi * 8;
      out[(size_t)mrow * HH + ncol] = (float)acc[t][r] * sc + bv;
    }
  }
}

__global__ __launch_bounds__(256) void gemm_scores_k(
    const signed char* __restrict__ Qq, const signed char* __restrict__ Kq,
    const float* __restrict__ mask, const float* __restrict__ scal,
    float* __restrict__ scores) {
  const int wid  = blockIdx.x * (blockDim.x >> 5) + (threadIdx.x >> 5);
  const int lane = threadIdx.x & 31;
  const int bh = wid >> 10;
  if (bh >= BB * NHH) return;
  const int rem = wid & 1023;
  const int tm = rem >> 4, sn = rem & 15;
  const int b = bh >> 4, h = bh & 15;

  const signed char* abase =
      Qq + (size_t)(b * SS + tm * 16) * HH + h * DHH;
  v8i a = load_a_frag(abase, HH, lane);
  v8i acc[4] = {};
#pragma unroll
  for (int t = 0; t < 4; ++t) {
    const signed char* bbase =
        Kq + (size_t)(b * SS + sn * 64 + t * 16) * HH + h * DHH;
    v8i bf = load_b_frag(bbase, HH, lane);
    acc[t] = __builtin_amdgcn_wmma_i32_16x16x64_iu8(true, a, true, bf, acc[t],
                                                    false, false);
  }
  const float sc = scal[16 + 4] * scal[16 + 5] * 0.125f;
  const int hi = lane >> 4, nloc = lane & 15;
#pragma unroll
  for (int t = 0; t < 4; ++t) {
    const int ncol = sn * 64 + t * 16 + nloc;
    const float mk = mask[b * SS + ncol];
#pragma unroll
    for (int r = 0; r < 8; ++r) {
      const int mrow = tm * 16 + r + hi * 8;
      scores[((size_t)bh * SS + mrow) * SS + ncol] = (float)acc[t][r] * sc + mk;
    }
  }
}

__global__ __launch_bounds__(256) void softmax_k(const float* __restrict__ sc,
                                                 float* __restrict__ pr) {
  __shared__ float red[256];
  const size_t row = blockIdx.x;
  const float* s = sc + row * SS;
  float* p = pr + row * SS;
  const int t = threadIdx.x;
  float v0 = s[t], v1 = s[t + 256], v2 = s[t + 512], v3 = s[t + 768];
  red[t] = fmaxf(fmaxf(v0, v1), fmaxf(v2, v3));
  __syncthreads();
  for (int o = 128; o > 0; o >>= 1) {
    if (t < o) red[t] = fmaxf(red[t], red[t + o]);
    __syncthreads();
  }
  const float rmax = red[0];
  __syncthreads();
  float e0 = expf(v0 - rmax), e1 = expf(v1 - rmax);
  float e2 = expf(v2 - rmax), e3 = expf(v3 - rmax);
  red[t] = e0 + e1 + e2 + e3;
  __syncthreads();
  for (int o = 128; o > 0; o >>= 1) {
    if (t < o) red[t] += red[t + o];
    __syncthreads();
  }
  const float inv = 1.0f / red[0];
  p[t] = e0 * inv; p[t + 256] = e1 * inv;
  p[t + 512] = e2 * inv; p[t + 768] = e3 * inv;
}

__global__ __launch_bounds__(256) void gemm_pv_k(
    const float* __restrict__ probs, const signed char* __restrict__ VqT,
    const float* __restrict__ scal, float* __restrict__ ctx2,
    float* __restrict__ ctx) {
  const int wid  = blockIdx.x * (blockDim.x >> 5) + (threadIdx.x >> 5);
  const int lane = threadIdx.x & 31;
  const int bh = wid >> 6;
  if (bh >= BB * NHH) return;
  const int tm = wid & 63;
  const int b = bh >> 4, h = bh & 15;

  const float sp = scal[8 + 7];
  const float* arow = probs + (size_t)bh * SS * SS + (size_t)(tm * 16) * SS;
  const signed char* bbase = VqT + (size_t)bh * (DHH * SS);

  v8i acc[4] = {};
  v8i aA = load_a_frag_quant(arow, SS, lane, sp);
  v8i bA[4], bB[4];
#pragma unroll
  for (int t = 0; t < 4; ++t)
    bA[t] = load_b_frag(bbase + (size_t)(t * 16) * SS, SS, lane);

#pragma unroll 1
  for (int kb = 0; kb < SS; kb += 128) {
    v8i aB = load_a_frag_quant(arow + kb + 64, SS, lane, sp);
#pragma unroll
    for (int t = 0; t < 4; ++t)
      bB[t] = load_b_frag(bbase + (size_t)(t * 16) * SS + kb + 64, SS, lane);
#pragma unroll
    for (int t = 0; t < 4; ++t)
      acc[t] = __builtin_amdgcn_wmma_i32_16x16x64_iu8(true, aA, true, bA[t],
                                                      acc[t], false, false);
    const int kn = (kb + 128 < SS) ? (kb + 128) : kb;
    aA = load_a_frag_quant(arow + kn, SS, lane, sp);
#pragma unroll
    for (int t = 0; t < 4; ++t)
      bA[t] = load_b_frag(bbase + (size_t)(t * 16) * SS + kn, SS, lane);
#pragma unroll
    for (int t = 0; t < 4; ++t)
      acc[t] = __builtin_amdgcn_wmma_i32_16x16x64_iu8(true, aB, true, bB[t],
                                                      acc[t], false, false);
  }

  const float scv = scal[16 + 7] * scal[16 + 6];
  const int hi = lane >> 4, nloc = lane & 15;
#pragma unroll
  for (int t = 0; t < 4; ++t) {
    const int ncol = t * 16 + nloc;
#pragma unroll
    for (int r = 0; r < 8; ++r) {
      const int mrow = tm * 16 + r + hi * 8;
      const float val = (float)acc[t][r] * scv;
      ctx2[((size_t)bh * SS + mrow) * DHH + ncol] = val;
      ctx[((size_t)(b * SS + mrow)) * HH + h * DHH + ncol] = val;
    }
  }
}

extern "C" void kernel_launch(void* const* d_in, const int* in_sizes, int n_in,
                              void* d_out, int out_size, void* d_ws,
                              size_t ws_size, hipStream_t stream) {
  (void)in_sizes; (void)n_in; (void)out_size; (void)ws_size;
  const float* hs   = (const float*)d_in[0];
  const float* mask = (const float*)d_in[1];
  const float* Wq   = (const float*)d_in[2];
  const float* bq   = (const float*)d_in[3];
  const float* Wk   = (const float*)d_in[4];
  const float* bk   = (const float*)d_in[5];
  const float* Wv   = (const float*)d_in[6];
  const float* bv   = (const float*)d_in[7];

  float* out = (float*)d_out;
  float* out_ctx  = out;
  float* out_sc   = out + (size_t)4194304;
  float* out_pr   = out + (size_t)71303168;
  float* out_ctx2 = out + (size_t)138412032;
  float* out_vq   = out + (size_t)142606336;

  char* ws = (char*)d_ws;
  size_t off = 0;
  auto take = [&](size_t bytes) -> char* {
    char* p = ws + off; off += (bytes + 255) & ~(size_t)255; return p;
  };
  float*       scal = (float*)take(1024);
  signed char* xq   = (signed char*)take((size_t)MR * HH);
  signed char* wqq  = (signed char*)take((size_t)HH * HH);
  signed char* wkq  = (signed char*)take((size_t)HH * HH);
  signed char* wvq  = (signed char*)take((size_t)HH * HH);
  float*       qf   = (float*)take((size_t)MR * HH * 4);
  float*       kf   = (float*)take((size_t)MR * HH * 4);
  float*       vf   = (float*)take((size_t)MR * HH * 4);
  signed char* qq   = (signed char*)take((size_t)MR * HH);
  signed char* kq   = (signed char*)take((size_t)MR * HH);
  signed char* vqT  = (signed char*)take((size_t)MR * HH);

  const int nHid = MR * HH;
  const int nW   = HH * HH;
  const int nPr  = BB * NHH * SS * SS;
  unsigned* am = (unsigned*)scal;

  init_scal_k<<<1, 32, 0, stream>>>(scal);
  absmax_k<<<1024, 256, 0, stream>>>(hs, nHid, am + 0);
  absmax_k<<<512, 256, 0, stream>>>(Wq, nW, am + 1);
  absmax_k<<<512, 256, 0, stream>>>(Wk, nW, am + 2);
  absmax_k<<<512, 256, 0, stream>>>(Wv, nW, am + 3);
  finalize_scales_k<<<1, 8, 0, stream>>>(scal, 0, 4);
  quant8_pack4_k<<<nHid / 4 / 256, 256, 0, stream>>>(hs, xq, scal, 0, nHid / 4);
  quant8_pack4_k<<<nW / 4 / 256, 256, 0, stream>>>(Wq, wqq, scal, 1, nW / 4);
  quant8_pack4_k<<<nW / 4 / 256, 256, 0, stream>>>(Wk, wkq, scal, 2, nW / 4);
  quant8_pack4_k<<<nW / 4 / 256, 256, 0, stream>>>(Wv, wvq, scal, 3, nW / 4);

  gemm_qkv_k<<<512, 256, 0, stream>>>(xq, wqq, bq, scal, 1, qf);
  gemm_qkv_k<<<512, 256, 0, stream>>>(xq, wkq, bk, scal, 2, kf);
  gemm_qkv_k<<<512, 256, 0, stream>>>(xq, wvq, bv, scal, 3, vf);

  absmax_k<<<1024, 256, 0, stream>>>(qf, nHid, am + 4);
  absmax_k<<<1024, 256, 0, stream>>>(kf, nHid, am + 5);
  absmax_k<<<1024, 256, 0, stream>>>(vf, nHid, am + 6);
  finalize_scales_k<<<1, 8, 0, stream>>>(scal, 4, 3);
  quant8_pack4_k<<<nHid / 4 / 256, 256, 0, stream>>>(qf, qq, scal, 4, nHid / 4);
  quant8_pack4_k<<<nHid / 4 / 256, 256, 0, stream>>>(kf, kq, scal, 5, nHid / 4);
  quant_v_k<<<nHid / 256, 256, 0, stream>>>(vf, vqT, out_vq, scal, nHid);

  gemm_scores_k<<<8192, 256, 0, stream>>>(qq, kq, mask, scal, out_sc);
  softmax_k<<<BB * NHH * SS, 256, 0, stream>>>(out_sc, out_pr);

  absmax_k<<<2048, 256, 0, stream>>>(out_pr, nPr, am + 7);
  finalize_scales_k<<<1, 8, 0, stream>>>(scal, 7, 1);
  gemm_pv_k<<<512, 256, 0, stream>>>(out_pr, vqT, scal, out_ctx2, out_ctx);
}
